// Attention_53197464928802
// MI455X (gfx1250) — matrix operand order repaired and checked by reading, not compiled, not run on hardware
//
#include <hip/hip_runtime.h>

// ---------------- CDNA5 WMMA types ----------------
typedef __attribute__((ext_vector_type(16))) __bf16 v16bf;
typedef __attribute__((ext_vector_type(8)))  float  v8f;

#define NB    8     // batch
#define NC    512   // channels
#define NO    1536  // 3*C
#define NH    8     // heads
#define DH    64    // dim per head
#define HWD   1024  // 32*32
#define WDIM  32

// ---------------- async global->LDS (guarded; falls back to plain copy) ----
#if __has_builtin(__builtin_amdgcn_global_load_async_to_lds_b128) && \
    __has_builtin(__builtin_amdgcn_s_wait_asynccnt)
#define HAVE_ASYNC_LDS 1
#else
#define HAVE_ASYNC_LDS 0
#endif

#if HAVE_ASYNC_LDS
typedef __attribute__((__vector_size__(4 * sizeof(int)))) int v4i_t;
typedef __attribute__((address_space(1))) v4i_t* g_v4p;   // global int4*
typedef __attribute__((address_space(3))) v4i_t* l_v4p;   // LDS int4*
__device__ __forceinline__ void async_b128(const void* g, void* l) {
    // per-lane: LDS[l .. l+15] = MEM[g .. g+15], tracked by ASYNCcnt
    __builtin_amdgcn_global_load_async_to_lds_b128(
        (g_v4p)(unsigned long long)g,
        (l_v4p)(unsigned int)(unsigned long long)l, 0, 0);
}
#endif

__device__ __forceinline__ unsigned short f32_to_bf16(float f) {
    unsigned int u = __float_as_uint(f);
    unsigned int r = (u + 0x7FFFu + ((u >> 16) & 1u)) >> 16;   // RNE
    return (unsigned short)r;
}
__device__ __forceinline__ unsigned int pack_bf16x2(float a, float b) {
    return (unsigned int)f32_to_bf16(a) | ((unsigned int)f32_to_bf16(b) << 16);
}

// Fragment builders from a row pointer (K contiguous in memory).
// A layout: lanes 0-15 -> M=lane, K in {0..7,16..23}; lanes 16-31 -> M=lane-16, K in {8..15,24..31}
// B layout: lanes 0-15 -> N=lane, K in {0..7,16..23}; lanes 16-31 -> N=lane-16, K in {8..15,24..31}
__device__ __forceinline__ v16bf load_frag_a(const unsigned short* rowp, int hi) {
    union { v16bf v; unsigned int u[8]; } r;
#pragma unroll
    for (int q = 0; q < 8; ++q) {
        int k0 = 2*q + (q >= 4 ? 8 : 0) + (hi ? 8 : 0);
        r.u[q] = *(const unsigned int*)(rowp + k0);
    }
    return r.v;
}
__device__ __forceinline__ v16bf load_frag_b(const unsigned short* rowp, int hi) {
    union { v16bf v; unsigned int u[8]; } r;
#pragma unroll
    for (int q = 0; q < 8; ++q) {
        int k0 = 2*q + (q >= 4 ? 8 : 0) + (hi ? 8 : 0);
        r.u[q] = *(const unsigned int*)(rowp + k0);
    }
    return r.v;
}
__device__ __forceinline__ v8f wmma_bf16(v16bf a, v16bf b, v8f c) {
    return __builtin_amdgcn_wmma_f32_16x16x32_bf16(false, a, false, b, (short)0, c, false, false);
}

// =====================================================================
// K1: QKV projection.  Out[o,pos] = sum_c W[o,c]*fmap[b,c,pos]
// Block: 256 thr (8 waves), tile M=64(o) x N=32(pos), Kstep=32.
// =====================================================================
__global__ __launch_bounds__(256) void qkv_gemm(
    const float* __restrict__ fmap, const float* __restrict__ wqkv,
    unsigned short* __restrict__ Qbf, unsigned short* __restrict__ Kbf,
    unsigned short* __restrict__ Vbf)
{
    __shared__ unsigned short ldsA[64 * 32];   // [m][k]
    __shared__ unsigned short ldsBT[32 * 34];  // [n][k], stride 34

    const int b  = blockIdx.z;
    const int o0 = blockIdx.x * 64;
    const int p0 = blockIdx.y * 32;
    const int tid  = threadIdx.x;
    const int lane = tid & 31, wave = tid >> 5;
    const int hi = lane >> 4, l16 = lane & 15;
    const int wm = wave >> 1, wn = wave & 1;

    const float* X = fmap + (size_t)b * NC * HWD;

    v8f acc = {};
    for (int ks = 0; ks < NC; ks += 32) {
        {   // stage A: W[o0+m][ks+k], f32 -> bf16
            int m  = tid >> 2;
            int c0 = (tid & 3) * 8;
            const float* src = wqkv + (size_t)(o0 + m) * NC + ks + c0;
            unsigned int* dst = (unsigned int*)(ldsA + m * 32 + c0);
#pragma unroll
            for (int q = 0; q < 4; ++q)
                dst[q] = pack_bf16x2(src[2*q], src[2*q + 1]);
        }
        {   // stage B transposed: fmap[b][ks+k][p0+n] -> ldsBT[n][k]
            int k  = tid >> 3;
            int n0 = (tid & 7) * 4;
            const float* src = X + (size_t)(ks + k) * HWD + p0 + n0;
#pragma unroll
            for (int q = 0; q < 4; ++q)
                ldsBT[(n0 + q) * 34 + k] = f32_to_bf16(src[q]);
        }
        __syncthreads();
        v16bf af = load_frag_a(ldsA  + (wm * 16 + l16) * 32, hi);
        v16bf bf = load_frag_b(ldsBT + (wn * 16 + l16) * 34, hi);
        acc = wmma_bf16(af, bf, acc);
        __syncthreads();
    }

    const int n   = wn * 16 + l16;
    const int pos = p0 + n;
#pragma unroll
    for (int r = 0; r < 8; ++r) {
        int m = r + hi * 8;
        int o = o0 + wm * 16 + m;
        float v = acc[r];
        if (o < 512) {
            int head = o >> 6, dd = o & 63;
            Qbf[(((size_t)(b * NH + head)) * HWD + pos) * DH + dd] = f32_to_bf16(v * 0.125f);
        } else if (o < 1024) {
            int o2 = o - 512; int head = o2 >> 6, dd = o2 & 63;
            Kbf[(((size_t)(b * NH + head)) * HWD + pos) * DH + dd] = f32_to_bf16(v);
        } else {
            int o3 = o - 1024; int head = o3 >> 6, dd = o3 & 63;
            Vbf[(((size_t)(b * NH + head)) * HWD + pos) * DH + dd] = f32_to_bf16(v);
        }
    }
}

// =====================================================================
// K2: relative-logit tables.  RW[i][m] = sum_d Q[i][d]*rel_w[m][d]
// =====================================================================
__global__ __launch_bounds__(256) void rel_gemm(
    const unsigned short* __restrict__ Qbf,
    const float* __restrict__ relw, const float* __restrict__ relh,
    float* __restrict__ RW, float* __restrict__ RH)
{
    __shared__ unsigned short ldsR[64 * 64];  // [m][d] bf16, row 63 zero

    const int mb = blockIdx.x;
    const int bh = blockIdx.y;
    const float* rel = blockIdx.z ? relh : relw;
    float* out       = blockIdx.z ? RH   : RW;
    const int tid = threadIdx.x;

    for (int e = tid; e < 64 * 64; e += 256) {
        int m = e >> 6, d = e & 63;
        float f = (m < 63) ? rel[m * 64 + d] : 0.f;
        ldsR[e] = f32_to_bf16(f);
    }
    __syncthreads();

    const int lane = tid & 31, wave = tid >> 5;
    const int hi = lane >> 4, l16 = lane & 15;
    const size_t row0 = (size_t)bh * HWD + mb * 128 + wave * 16;

    const unsigned short* qrow = Qbf + (row0 + l16) * DH;
    v16bf a0 = load_frag_a(qrow,      hi);
    v16bf a1 = load_frag_a(qrow + 32, hi);

#pragma unroll
    for (int nt = 0; nt < 4; ++nt) {
        v8f acc = {};
        const unsigned short* rrow = ldsR + (nt * 16 + l16) * 64;
        acc = wmma_bf16(a0, load_frag_b(rrow,      hi), acc);
        acc = wmma_bf16(a1, load_frag_b(rrow + 32, hi), acc);
#pragma unroll
        for (int r = 0; r < 8; ++r) {
            int m = r + hi * 8;
            out[(row0 + m) * 64 + nt * 16 + l16] = acc[r];
        }
    }
}

// =====================================================================
// K3: sim = q.kT + pos.  64x64 tile per block per (b,head).
// Staging uses GLOBAL_LOAD_ASYNC_TO_LDS_B128 when available (ASYNCcnt).
// =====================================================================
__global__ __launch_bounds__(256) void sim_kernel(
    const unsigned short* __restrict__ Qbf, const unsigned short* __restrict__ Kbf,
    const float* __restrict__ RW, const float* __restrict__ RH,
    float* __restrict__ simout)
{
    __shared__ unsigned short ldsQ[64 * 64];
    __shared__ unsigned short ldsK[64 * 64];
    __shared__ float ldsRW[64 * 64];
    __shared__ float ldsRH[64 * 64];

    const int bh = blockIdx.z;
    const int i0 = blockIdx.y * 64;
    const int j0 = blockIdx.x * 64;
    const int tid = threadIdx.x;

    const char* gq = (const char*)(Qbf + ((size_t)bh * HWD + i0) * DH);   // 8 KB
    const char* gk = (const char*)(Kbf + ((size_t)bh * HWD + j0) * DH);   // 8 KB
    const char* gw = (const char*)(RW  + ((size_t)bh * HWD + i0) * 64);   // 16 KB
    const char* gh = (const char*)(RH  + ((size_t)bh * HWD + i0) * 64);   // 16 KB

#if HAVE_ASYNC_LDS
    {
#pragma unroll
        for (int rpt = 0; rpt < 2; ++rpt) {
            int off = (rpt * 256 + tid) * 16;
            async_b128(gq + off, (char*)ldsQ + off);
            async_b128(gk + off, (char*)ldsK + off);
        }
#pragma unroll
        for (int rpt = 0; rpt < 4; ++rpt) {
            int off = (rpt * 256 + tid) * 16;
            async_b128(gw + off, (char*)ldsRW + off);
            async_b128(gh + off, (char*)ldsRH + off);
        }
        __builtin_amdgcn_s_wait_asynccnt(0);
    }
#else
    {
        const unsigned int* qs = (const unsigned int*)gq;
        const unsigned int* ks = (const unsigned int*)gk;
        unsigned int* qd = (unsigned int*)ldsQ;
        unsigned int* kd = (unsigned int*)ldsK;
        for (int e = tid; e < 2048; e += 256) { qd[e] = qs[e]; kd[e] = ks[e]; }
        const float4* rws = (const float4*)gw;
        const float4* rhs = (const float4*)gh;
        float4* rwd = (float4*)ldsRW;
        float4* rhd = (float4*)ldsRH;
        for (int e = tid; e < 1024; e += 256) { rwd[e] = rws[e]; rhd[e] = rhs[e]; }
    }
#endif
    __syncthreads();

    const int lane = tid & 31, wave = tid >> 5;
    const int hi = lane >> 4, l16 = lane & 15;
    const int wm = wave >> 1;   // 0..3
    const int wn2 = wave & 1;   // 0..1  (two 16-col tiles each)

    const unsigned short* qrow = ldsQ + (wm * 16 + l16) * 64;
    v16bf a0 = load_frag_a(qrow,      hi);
    v16bf a1 = load_frag_a(qrow + 32, hi);

#pragma unroll
    for (int t = 0; t < 2; ++t) {
        const int nloc = wn2 * 32 + t * 16 + l16;
        const unsigned short* krow = ldsK + nloc * 64;
        v8f acc = {};
        acc = wmma_bf16(a0, load_frag_b(krow,      hi), acc);
        acc = wmma_bf16(a1, load_frag_b(krow + 32, hi), acc);
        const int j = j0 + nloc;
#pragma unroll
        for (int r = 0; r < 8; ++r) {
            int m = r + hi * 8;
            int iloc = wm * 16 + m;
            int i = i0 + iloc;
            int yd = (j & 31) - (i & 31) + 31;
            int xd = (j >> 5) - (i >> 5) + 31;
            float val = acc[r] + ldsRW[iloc * 64 + yd] + ldsRH[iloc * 64 + xd];
            simout[((size_t)bh * HWD + i) * HWD + j] = val;
        }
    }
}

// =====================================================================
// K4: fused softmax + attn.V + output scatter.  32 rows per block,
// K-step 64 -> 2 WMMAs per barrier pair.
// =====================================================================
__global__ __launch_bounds__(256) void softmax_av(
    const float* __restrict__ sim, const unsigned short* __restrict__ Vbf,
    float* __restrict__ out)
{
    __shared__ float redbuf[32 * 8];
    __shared__ float rowmax[32];
    __shared__ float rowinv[32];
    __shared__ unsigned short ldsP[32 * 64];    // exp(sim - max) bf16, [m][k]
    __shared__ unsigned short ldsVT[64 * 66];   // [dd][k], stride 66

    const int bh = blockIdx.y;
    const int i0 = blockIdx.x * 32;
    const int tid = threadIdx.x;

    // ---- phase 1: row max / sum of exp over 1024 cols (8 threads per row)
    const int rr = tid >> 3, cch = tid & 7;
    const float* simrow = sim + ((size_t)bh * HWD + i0 + rr) * HWD + cch * 128;
    float mx = -1e30f;
    for (int q = 0; q < 128; ++q) mx = fmaxf(mx, simrow[q]);
    redbuf[rr * 8 + cch] = mx;
    __syncthreads();
    if (cch == 0) {
        float m2 = redbuf[rr * 8];
#pragma unroll
        for (int q = 1; q < 8; ++q) m2 = fmaxf(m2, redbuf[rr * 8 + q]);
        rowmax[rr] = m2;
    }
    __syncthreads();
    const float rm = rowmax[rr];
    float s = 0.f;
    for (int q = 0; q < 128; ++q) s += __expf(simrow[q] - rm);
    redbuf[rr * 8 + cch] = s;
    __syncthreads();
    if (cch == 0) {
        float s2 = 0.f;
#pragma unroll
        for (int q = 0; q < 8; ++q) s2 += redbuf[rr * 8 + q];
        rowinv[rr] = 1.f / s2;
    }
    __syncthreads();

    // ---- phase 2: exp(sim-max) @ V via WMMA, K loop over 1024 in 64-steps
    const int lane = tid & 31, wave = tid >> 5;
    const int hi = lane >> 4, l16 = lane & 15;
    const int wm = wave & 1;        // 2 row tiles
    const int wn = wave >> 1;       // 4 dd tiles

    v8f acc = {};
    for (int ks = 0; ks < HWD; ks += 64) {
        {   // stage P = exp(sim - rowmax) as bf16, 32 x 64
            int pr = tid >> 3, pc0 = (tid & 7) * 8;
            const float* src = sim + ((size_t)bh * HWD + i0 + pr) * HWD + ks + pc0;
            float prm = rowmax[pr];
            unsigned int* dst = (unsigned int*)(ldsP + pr * 64 + pc0);
            dst[0] = pack_bf16x2(__expf(src[0] - prm), __expf(src[1] - prm));
            dst[1] = pack_bf16x2(__expf(src[2] - prm), __expf(src[3] - prm));
            dst[2] = pack_bf16x2(__expf(src[4] - prm), __expf(src[5] - prm));
            dst[3] = pack_bf16x2(__expf(src[6] - prm), __expf(src[7] - prm));
        }
        {   // stage V transposed: Vbf[bh][ks+k][dd] -> ldsVT[dd][k], 64 x 64
#pragma unroll
            for (int rpt = 0; rpt < 2; ++rpt) {
                int k  = rpt * 32 + (tid >> 3);
                int d0 = (tid & 7) * 8;
                const unsigned short* src = Vbf + ((size_t)bh * HWD + ks + k) * DH + d0;
#pragma unroll
                for (int q = 0; q < 8; ++q) ldsVT[(d0 + q) * 66 + k] = src[q];
            }
        }
        __syncthreads();
        const unsigned short* prow = ldsP  + (wm * 16 + l16) * 64;
        const unsigned short* vrow = ldsVT + (wn * 16 + l16) * 66;
        acc = wmma_bf16(load_frag_a(prow,      hi), load_frag_b(vrow,      hi), acc);
        acc = wmma_bf16(load_frag_a(prow + 32, hi), load_frag_b(vrow + 32, hi), acc);
        __syncthreads();
    }

    // ---- epilogue: scale by 1/rowsum, scatter to [b, head*64+dd, pos]
    const int n = wn * 16 + l16;            // dd
    const int b = bh >> 3, head = bh & 7;
#pragma unroll
    for (int r = 0; r < 8; ++r) {
        int m = r + hi * 8;
        int iloc = wm * 16 + m;
        int i = i0 + iloc;
        float val = acc[r] * rowinv[iloc];
        int ch = head * 64 + n;
        out[((size_t)b * NC + ch) * HWD + i] = val;
    }
}

// =====================================================================
extern "C" void kernel_launch(void* const* d_in, const int* in_sizes, int n_in,
                              void* d_out, int out_size, void* d_ws, size_t ws_size,
                              hipStream_t stream)
{
    const float* fmap = (const float*)d_in[0];   // [8,512,32,32]
    const float* wqkv = (const float*)d_in[1];   // [1536,512]
    const float* relw = (const float*)d_in[2];   // [63,64]
    const float* relh = (const float*)d_in[3];   // [63,64]

    float* out0   = (float*)d_out;                 // [8,512,32,32]
    float* simout = out0 + (size_t)NB * NC * HWD;  // [8,8,1024,1024]

    char* ws = (char*)d_ws;
    const size_t qkvElems = (size_t)NB * NH * HWD * DH;      // 4,194,304
    unsigned short* Qbf = (unsigned short*)ws;
    unsigned short* Kbf = Qbf + qkvElems;
    unsigned short* Vbf = Kbf + qkvElems;
    float* RW = (float*)(ws + 3 * qkvElems * sizeof(unsigned short));
    float* RH = RW + qkvElems;

    dim3 g1(NO / 64, HWD / 32, NB);            // 24 x 32 x 8
    qkv_gemm<<<g1, 256, 0, stream>>>(fmap, wqkv, Qbf, Kbf, Vbf);

    dim3 g2(HWD / 128, NB * NH, 2);            // 8 x 64 x 2
    rel_gemm<<<g2, 256, 0, stream>>>(Qbf, relw, relh, RW, RH);

    dim3 g3(HWD / 64, HWD / 64, NB * NH);      // 16 x 16 x 64
    sim_kernel<<<g3, 256, 0, stream>>>(Qbf, Kbf, RW, RH, simout);

    dim3 g4(HWD / 32, NB * NH);                // 32 x 64
    softmax_av<<<g4, 256, 0, stream>>>(simout, Vbf, out0);
}
